// Qwen3Attention_2628519985676
// MI455X (gfx1250) — matrix operand order repaired and checked by reading, not compiled, not run on hardware
//
#include <hip/hip_runtime.h>
#include <hip/hip_bf16.h>

// ---------------------------------------------------------------------------
// Qwen3 attention block for MI455X (gfx1250, wave32, WMMA).
// All matmuls use V_WMMA_F32_16X16X32_BF16 (bf16 operands, fp32 accumulate).
// GEMM: 64x64 tile per wave (16 WMMAs / 8 b128-load-pairs per K-step) for
// 2x the L2 arithmetic intensity of a 32x32 tile; global_prefetch one
// K-step ahead.
// ---------------------------------------------------------------------------

typedef __attribute__((ext_vector_type(16))) __bf16 bf16x16;
typedef __attribute__((ext_vector_type(8)))  __bf16 bf16x8;
typedef __attribute__((ext_vector_type(8)))  float  f32x8;

#define T_SEQ  2048
#define HIDDEN 2048
#define NH     16
#define NKV    8
#define HD     128
#define QSZ    (NH * HD)          // 2048
#define KVSZ   (NKV * HD)         // 1024
#define QKV_N  (QSZ + 2 * KVSZ)   // 4096

// ---- fragment loaders (contiguous k order k = 16*(lane/16) + s, wave32) -------------------------
// A-matrix 16x32 bf16: lane m (0-15) holds row m, K k0..k0+15;
// lane m+16 holds row m, K k0+16..k0+31.
__device__ __forceinline__ bf16x16 load_a_frag(const __bf16* A, int lda,
                                               int m0, int k0, int lane) {
  int m    = m0 + (lane & 15);
  int koff = (lane >> 4) << 4;
  const __bf16* p = A + (size_t)m * lda + (k0 + koff);
  bf16x8 lo = *(const bf16x8*)p;
  bf16x8 hi = *(const bf16x8*)(p + 8);
  return __builtin_shufflevector(lo, hi, 0, 1, 2, 3, 4, 5, 6, 7,
                                 8, 9, 10, 11, 12, 13, 14, 15);
}

// B-matrix 32x16 bf16 where B[k][n] = W[n][k] (NT GEMM): lane n (0-15) holds
// column n with K k0..k0+15; lane n+16 holds K k0+16..k0+31. Contiguous 32B.
__device__ __forceinline__ bf16x16 load_b_frag(const __bf16* W, int ldw,
                                               int n0, int k0, int lane) {
  int n  = n0 + (lane & 15);
  int kk = k0 + ((lane >> 4) << 4);
  return *(const bf16x16*)(W + (size_t)n * ldw + kk);
}

// ---- fp32 -> bf16 elementwise convert --------------------------------------
__global__ void cvt_f32_bf16(const float* __restrict__ src,
                             __bf16* __restrict__ dst, int n) {
  int i = blockIdx.x * 256 + threadIdx.x;
  if (i < n) dst[i] = (__bf16)src[i];
}

// ---- NT GEMM: C[M][N] = sum_k A[m][k] * W[n][k] ----------------------------
// One 64x64 output tile per wave, 4 waves per block.
__global__ void __launch_bounds__(128)
gemm_bf16_nt(const __bf16* __restrict__ A, const __bf16* __restrict__ W,
             float* __restrict__ C, int M, int N, int K) {
  const int lane = threadIdx.x & 31;
  const int wid  = (blockIdx.x << 2) | (threadIdx.x >> 5);
  const int tilesN = N >> 6;
  const int tm = (wid / tilesN) << 6;
  const int tn = (wid % tilesN) << 6;
  if (tm >= M) return;  // wave-uniform

  f32x8 c[4][4] = {};
  for (int k0 = 0; k0 < K; k0 += 32) {
    bf16x16 a[4], b[4];
#pragma unroll
    for (int i = 0; i < 4; ++i) {
      a[i] = load_a_frag(A, K, tm + i * 16, k0, lane);
      b[i] = load_b_frag(W, K, tn + i * 16, k0, lane);
    }
    if (k0 + 32 < K) {  // wave-uniform: pull next K-step toward the WGP
      __builtin_prefetch(A + (size_t)(tm + lane) * K + (k0 + 32), 0, 3);
      __builtin_prefetch(W + (size_t)(tn + lane) * K + (k0 + 32), 0, 3);
    }
#pragma unroll
    for (int i = 0; i < 4; ++i)
#pragma unroll
      for (int j = 0; j < 4; ++j)
        c[i][j] = __builtin_amdgcn_wmma_f32_16x16x32_bf16(
            false, a[i], false, b[j], (short)0, c[i][j], false, false);
  }
  // C/D layout: VGPR r, lane l -> row = (l>>4)*8 + r, col = l&15
  const int rbase = (lane >> 4) << 3;
  const int col0  = lane & 15;
#pragma unroll
  for (int i = 0; i < 4; ++i)
#pragma unroll
    for (int j = 0; j < 4; ++j) {
      float* Cp = C + (size_t)(tm + i * 16 + rbase) * N + tn + j * 16 + col0;
#pragma unroll
      for (int r = 0; r < 8; ++r) Cp[(size_t)r * N] = c[i][j][r];
    }
}

// ---- RMSNorm + RoPE on Q/K, transpose-convert V ----------------------------
// grid = (T, 32): y in [0,16) q-head, [16,24) k-head, [24,32) v-head.
__global__ void __launch_bounds__(128)
normrope_kernel(const float* __restrict__ qkv, const int* __restrict__ positions,
                const float* __restrict__ qw, const float* __restrict__ kw,
                __bf16* __restrict__ Qh, __bf16* __restrict__ Kh,
                __bf16* __restrict__ Vt) {
  const int t = blockIdx.x;
  const int y = blockIdx.y;
  const int d = threadIdx.x;  // 0..127
  __shared__ float red[128];
  __shared__ float xs[128];
  const float* row = qkv + (size_t)t * QKV_N;

  if (y < NH + NKV) {  // uniform branch per block
    const bool isq = (y < NH);
    const int  h   = isq ? y : (y - NH);
    const float* w = isq ? qw : kw;
    const float  x = isq ? row[h * HD + d] : row[QSZ + h * HD + d];
    red[d] = x * x;
    __syncthreads();
#pragma unroll
    for (int s = 64; s > 0; s >>= 1) {
      if (d < s) red[d] += red[d + s];
      __syncthreads();
    }
    const float var = red[0] * (1.0f / 128.0f);
    const float xn  = x * rsqrtf(var + 1e-6f) * w[d];
    xs[d] = xn;
    __syncthreads();
    // RoPE
    const int   p    = positions[t];
    const int   i    = d & 63;
    const float invf = __expf(-(float)i * (9.210340371976184f / 64.0f));
    const float ang  = (float)p * invf;
    const float c = __cosf(ang), sn = __sinf(ang);
    float o;
    if (d < 64) o = xs[d] * c - xs[d + 64] * sn;
    else        o = xs[d] * c + xs[d - 64] * sn;
    __bf16* dst = isq ? (Qh + ((size_t)h * T_SEQ + t) * HD)
                      : (Kh + ((size_t)h * T_SEQ + t) * HD);
    dst[d] = (__bf16)o;
  } else {
    const int h = y - NH - NKV;
    const float x = row[QSZ + KVSZ + h * HD + d];
    Vt[((size_t)h * HD + d) * T_SEQ + t] = (__bf16)x;  // V transposed [d][t]
  }
}

// ---- Flash attention: one wave per (q-head, 32-query block) ----------------
__global__ void __launch_bounds__(32)
attn_kernel(const __bf16* __restrict__ Qh, const __bf16* __restrict__ Kh,
            const __bf16* __restrict__ Vt, __bf16* __restrict__ Ctx) {
  const int lane = threadIdx.x;
  const int qb   = blockIdx.x;   // 0..63
  const int h    = blockIdx.y;   // 0..15
  const int kvh  = h >> 1;       // G = 2
  const __bf16* Q  = Qh + (size_t)h   * T_SEQ * HD;
  const __bf16* Kp = Kh + (size_t)kvh * T_SEQ * HD;
  const __bf16* Vp = Vt + (size_t)kvh * HD * T_SEQ;
  const int qr0 = qb << 5;
  __shared__ __bf16 Plds[32 * 32];

  // Preload Q fragments: 2 (M tiles) x 4 (K steps of 32 over HD=128)
  bf16x16 qf[2][4];
#pragma unroll
  for (int mi = 0; mi < 2; ++mi)
#pragma unroll
    for (int ks = 0; ks < 4; ++ks)
      qf[mi][ks] = load_a_frag(Q, HD, qr0 + mi * 16, ks * 32, lane);

  f32x8 acc[2][8] = {};
  float mrow[2][8], lrow[2][8];
#pragma unroll
  for (int mi = 0; mi < 2; ++mi)
#pragma unroll
    for (int r = 0; r < 8; ++r) { mrow[mi][r] = -3.0e38f; lrow[mi][r] = 0.0f; }

  const float scale = 0.08838834764831845f;  // 128^-0.5
  const int rbase = (lane >> 4) << 3;
  const int nlane = lane & 15;

  for (int kb = 0; kb <= qb; ++kb) {
    const int kr0 = kb << 5;
    // S = Q K^T  (32x32 tile)
    f32x8 s[2][2] = {};
#pragma unroll
    for (int ks = 0; ks < 4; ++ks) {
      bf16x16 kf0 = load_b_frag(Kp, HD, kr0,      ks * 32, lane);
      bf16x16 kf1 = load_b_frag(Kp, HD, kr0 + 16, ks * 32, lane);
      s[0][0] = __builtin_amdgcn_wmma_f32_16x16x32_bf16(false, qf[0][ks], false, kf0,
                                                        (short)0, s[0][0], false, false);
      s[0][1] = __builtin_amdgcn_wmma_f32_16x16x32_bf16(false, qf[0][ks], false, kf1,
                                                        (short)0, s[0][1], false, false);
      s[1][0] = __builtin_amdgcn_wmma_f32_16x16x32_bf16(false, qf[1][ks], false, kf0,
                                                        (short)0, s[1][0], false, false);
      s[1][1] = __builtin_amdgcn_wmma_f32_16x16x32_bf16(false, qf[1][ks], false, kf1,
                                                        (short)0, s[1][1], false, false);
    }
    const bool diag = (kb == qb);
#pragma unroll
    for (int mi = 0; mi < 2; ++mi)
#pragma unroll
      for (int ni = 0; ni < 2; ++ni)
#pragma unroll
        for (int r = 0; r < 8; ++r) {
          float v = s[mi][ni][r] * scale;
          if (diag) {
            int rowi = qr0 + mi * 16 + rbase + r;
            int coli = kr0 + ni * 16 + nlane;
            if (coli > rowi) v = -3.0e38f;
          }
          s[mi][ni][r] = v;
        }
    // Online softmax; each row lives across one 16-lane half in a single VGPR.
#pragma unroll
    for (int mi = 0; mi < 2; ++mi)
#pragma unroll
      for (int r = 0; r < 8; ++r) {
        float mx = fmaxf(s[mi][0][r], s[mi][1][r]);
#pragma unroll
        for (int off = 1; off < 16; off <<= 1)
          mx = fmaxf(mx, __shfl_xor(mx, off, 32));
        const float mnew = fmaxf(mrow[mi][r], mx);
        const float corr = __expf(mrow[mi][r] - mnew);
        mrow[mi][r] = mnew;
        const float p0 = __expf(s[mi][0][r] - mnew);
        const float p1 = __expf(s[mi][1][r] - mnew);
        s[mi][0][r] = p0;
        s[mi][1][r] = p1;
        float ps = p0 + p1;
#pragma unroll
        for (int off = 1; off < 16; off <<= 1)
          ps += __shfl_xor(ps, off, 32);
        lrow[mi][r] = lrow[mi][r] * corr + ps;
#pragma unroll
        for (int nd = 0; nd < 8; ++nd) acc[mi][nd][r] *= corr;
      }
    // Stage P through LDS (C-layout -> A-layout reshuffle) as bf16
    __syncthreads();
#pragma unroll
    for (int mi = 0; mi < 2; ++mi)
#pragma unroll
      for (int ni = 0; ni < 2; ++ni)
#pragma unroll
        for (int r = 0; r < 8; ++r)
          Plds[(mi * 16 + rbase + r) * 32 + ni * 16 + nlane] =
              (__bf16)s[mi][ni][r];
    __syncthreads();
    bf16x16 pf[2];
#pragma unroll
    for (int mi = 0; mi < 2; ++mi) {
      const __bf16* pp = Plds + (mi * 16 + nlane) * 32 + ((lane >> 4) << 4);
      bf16x8 lo = *(const bf16x8*)pp;
      bf16x8 hi = *(const bf16x8*)(pp + 8);
      pf[mi] = __builtin_shufflevector(lo, hi, 0, 1, 2, 3, 4, 5, 6, 7,
                                       8, 9, 10, 11, 12, 13, 14, 15);
    }
    // ctx += P * V  (V^T layout makes B-fragments contiguous per lane)
#pragma unroll
    for (int nd = 0; nd < 8; ++nd) {
      bf16x16 vf = load_b_frag(Vp, T_SEQ, nd * 16, kr0, lane);
      acc[0][nd] = __builtin_amdgcn_wmma_f32_16x16x32_bf16(false, pf[0], false, vf,
                                                           (short)0, acc[0][nd], false, false);
      acc[1][nd] = __builtin_amdgcn_wmma_f32_16x16x32_bf16(false, pf[1], false, vf,
                                                           (short)0, acc[1][nd], false, false);
    }
    __syncthreads();
  }
  // Normalize and write ctx[t][h*128 + d] as bf16 for the O-projection GEMM
#pragma unroll
  for (int mi = 0; mi < 2; ++mi)
#pragma unroll
    for (int nd = 0; nd < 8; ++nd)
#pragma unroll
      for (int r = 0; r < 8; ++r) {
        const int t = qr0 + mi * 16 + rbase + r;
        const int d = nd * 16 + nlane;
        Ctx[(size_t)t * QSZ + h * HD + d] =
            (__bf16)(acc[mi][nd][r] / lrow[mi][r]);
      }
}

// ---------------------------------------------------------------------------
extern "C" void kernel_launch(void* const* d_in, const int* in_sizes, int n_in,
                              void* d_out, int out_size, void* d_ws, size_t ws_size,
                              hipStream_t stream) {
  const int*   positions = (const int*)d_in[0];
  const float* hidden    = (const float*)d_in[1];
  const float* qkv_w     = (const float*)d_in[2];
  const float* q_norm_w  = (const float*)d_in[3];
  const float* k_norm_w  = (const float*)d_in[4];
  const float* o_w       = (const float*)d_in[5];
  float* out = (float*)d_out;

  char* ws = (char*)d_ws;
  const size_t MiB = 1u << 20;
  // Workspace layout (needs 88 MiB):
  __bf16* hidden_bf = (__bf16*)(ws + 0 * MiB);    // 8 MiB  [2048][2048]
  __bf16* wqkv_bf   = (__bf16*)(ws + 8 * MiB);    // 16 MiB [4096][2048]
  __bf16* wo_bf     = (__bf16*)(ws + 24 * MiB);   // 8 MiB  [2048][2048]
  float*  qkv       = (float*) (ws + 32 * MiB);   // 32 MiB [2048][4096]
  __bf16* Qh        = (__bf16*)(ws + 64 * MiB);   // 8 MiB  [16][2048][128]
  __bf16* Kh        = (__bf16*)(ws + 72 * MiB);   // 4 MiB  [8][2048][128]
  __bf16* Vt        = (__bf16*)(ws + 76 * MiB);   // 4 MiB  [8][128][2048]
  __bf16* Ctx       = (__bf16*)(ws + 80 * MiB);   // 8 MiB  [2048][2048]

  // 1) fp32 -> bf16 conversions
  cvt_f32_bf16<<<(T_SEQ * HIDDEN + 255) / 256, 256, 0, stream>>>(
      hidden, hidden_bf, T_SEQ * HIDDEN);
  cvt_f32_bf16<<<(QKV_N * HIDDEN + 255) / 256, 256, 0, stream>>>(
      qkv_w, wqkv_bf, QKV_N * HIDDEN);
  cvt_f32_bf16<<<(HIDDEN * QSZ + 255) / 256, 256, 0, stream>>>(
      o_w, wo_bf, HIDDEN * QSZ);

  // 2) qkv = hidden @ qkv_w^T   (M=2048, N=4096, K=2048), 64x64 tiles
  gemm_bf16_nt<<<(T_SEQ / 64) * (QKV_N / 64) / 4, 128, 0, stream>>>(
      hidden_bf, wqkv_bf, qkv, T_SEQ, QKV_N, HIDDEN);

  // 3) RMSNorm + RoPE on Q/K; transpose-convert V
  normrope_kernel<<<dim3(T_SEQ, NH + 2 * NKV), 128, 0, stream>>>(
      qkv, positions, q_norm_w, k_norm_w, Qh, Kh, Vt);

  // 4) Flash attention
  attn_kernel<<<dim3(T_SEQ / 32, NH), 32, 0, stream>>>(Qh, Kh, Vt, Ctx);

  // 5) out = ctx @ o_w^T   (M=2048, N=2048, K=2048), 64x64 tiles
  gemm_bf16_nt<<<(T_SEQ / 64) * (HIDDEN / 64) / 4, 128, 0, stream>>>(
      Ctx, wo_bf, out, T_SEQ, HIDDEN, QSZ);
}
